// BiLSTM_CRF_78804059947429
// MI455X (gfx1250) — matrix operand order repaired and checked by reading, not compiled, not run on hardware
//
#include <hip/hip_runtime.h>

// ---------------- problem constants ----------------
constexpr int nB   = 64;
constexpr int nF   = 128;
constexpr int nT   = 1024;
constexpr int nH   = 256;
constexpr int nNT  = 11;
constexpr int nLBL = 4;
constexpr int nS   = nT * nLBL;   // 4096
constexpr int nG   = 4 * nH;      // 1024 gate columns
constexpr int nH2  = 2 * nH;      // 512
constexpr int nEMP = 48;          // padded emission cols (44 -> 48)

typedef __bf16 bf16_t;
typedef __attribute__((ext_vector_type(16))) __bf16 v16bf;
typedef __attribute__((ext_vector_type(8)))  float  v8f;
typedef __attribute__((ext_vector_type(4)))  float  v4f;
typedef __attribute__((ext_vector_type(4)))  int    v4i;

// CDNA5 async global->LDS path (ASYNCcnt), guarded so compile never regresses.
#if __has_builtin(__builtin_amdgcn_global_load_async_to_lds_b128)
#define HAVE_ASYNC_LDS 1
typedef __attribute__((address_space(1))) v4i* as1_v4i;   // global 128b payload ptr
typedef __attribute__((address_space(3))) v4i* as3_v4i;   // LDS 128b payload ptr
#else
#define HAVE_ASYNC_LDS 0
#endif

__device__ inline void wait_asynccnt0() {
#if HAVE_ASYNC_LDS
#if __has_builtin(__builtin_amdgcn_s_wait_asynccnt)
    __builtin_amdgcn_s_wait_asynccnt(0);
#else
    asm volatile("s_wait_asynccnt 0x0" ::: "memory");
#endif
#endif
}

// ---------------- helpers ----------------
__device__ inline bf16_t f2bf(float f) {
    unsigned u = __builtin_bit_cast(unsigned, f);
    u += 0x7FFFu + ((u >> 16) & 1u);          // round to nearest even
    unsigned short h = (unsigned short)(u >> 16);
    return __builtin_bit_cast(bf16_t, h);
}

__device__ inline float sigm(float x) { return 1.f / (1.f + __expf(-x)); }

__device__ inline v8f wmma_bf16(v16bf a, v16bf b, v8f c) {
    // D = A(16x32) * B(32x16) + C(16x16 f32), wave32
    return __builtin_amdgcn_wmma_f32_16x16x32_bf16(false, a, false, b, (short)0, c,
                                                   false, false);
}

// A fragment (16x32, bf16) from row-major source (stride ldk), contiguous k order (matches load_b_rm):
// lanes 0-15 hold K=[0..15] of row M=lane; lanes 16-31 hold K=[16..31].
__device__ inline v16bf load_a_rm(const bf16_t* src, int ldk, int kk, int lane) {
    const int m  = lane & 15;
    const int k0 = kk + ((lane >> 4) << 4);
    const bf16_t* p = src + (size_t)m * ldk;
    v16bf a;
#pragma unroll
    for (int e = 0; e < 16; ++e) { a[e] = p[k0 + e]; }
    return a;
}

// B fragment (32x16, bf16) where B(k,n) = W[n][k], W row-major with stride ldk.
// lanes 0-15: K=kk..kk+15 of column n; lanes 16-31: K=kk+16..kk+31. 32B contiguous load.
__device__ inline v16bf load_b_rm(const bf16_t* W, int ldk, int n, int kk, int lane) {
    const bf16_t* p = W + (size_t)n * ldk + kk + ((lane >> 4) << 4);
    return *reinterpret_cast<const v16bf*>(p);
}

// emission accessor: em[(b*nT + s/4)*48 + (s%4)*11 + j]
__device__ inline float emv(const float* em, int b, int s, int j) {
    return em[((size_t)b * nT + (s >> 2)) * nEMP + (s & 3) * nNT + j];
}

// ---------------- prep kernels ----------------
__global__ void k_transpose_x(const float* __restrict__ x, bf16_t* __restrict__ xt) {
    int idx = blockIdx.x * blockDim.x + threadIdx.x;   // over B*T*F
    if (idx >= nB * nT * nF) return;
    int f = idx % nF, t = (idx / nF) % nT, b = idx / (nF * nT);
    xt[idx] = f2bf(x[((size_t)b * nF + f) * nT + t]);
}

__global__ void k_cvt(const float* __restrict__ s, bf16_t* __restrict__ d, int n) {
    int i = blockIdx.x * blockDim.x + threadIdx.x;
    if (i < n) d[i] = f2bf(s[i]);
}

__global__ void k_cvt_pad_wout(const float* __restrict__ s, bf16_t* __restrict__ d) {
    int i = blockIdx.x * blockDim.x + threadIdx.x;     // over 48*512
    if (i >= nEMP * nH2) return;
    int row = i / nH2, col = i % nH2;
    d[i] = (row < nLBL * nNT) ? f2bf(s[row * nH2 + col]) : f2bf(0.f);
}

__global__ void k_rev(const int* __restrict__ lengths, int* __restrict__ rev) {
    int i = blockIdx.x * blockDim.x + threadIdx.x;     // over B*T
    if (i >= nB * nT) return;
    int t = i % nT, b = i / nT, len = lengths[b];
    rev[i] = (t < len) ? (len - 1 - t) : t;
}

// ---------------- input-projection GEMM: out[M][1024] = X[M][K] * W[1024][K]^T + bias
__global__ void k_xg_gemm(const bf16_t* __restrict__ X, const bf16_t* __restrict__ W,
                          const float* __restrict__ bias, float* __restrict__ out, int K) {
    extern __shared__ bf16_t sA[];                     // 16 x K tile
    const int tid = threadIdx.x, lane = tid & 31, wave = tid >> 5;   // 8 waves
    const int mbase = blockIdx.x * 16;
    const int nbase = (blockIdx.y * 8 + wave) * 16;
    for (int i = tid; i < 16 * K; i += 256)
        sA[i] = X[(size_t)(mbase + i / K) * K + (i % K)];
    __syncthreads();
    const int n = nbase + (lane & 15);
    v8f acc = {};
    for (int kk = 0; kk < K; kk += 32) {
        v16bf a = load_a_rm(sA, K, kk, lane);
        v16bf b = load_b_rm(W, K, n, kk, lane);
        acc = wmma_bf16(a, b, acc);
    }
    const float bv = bias[n];
    const int r0 = mbase + ((lane >> 4) << 3);
#pragma unroll
    for (int r = 0; r < 8; ++r)
        out[(size_t)(r0 + r) * nG + n] = acc[r] + bv;
}

// ---------------- persistent recurrent LSTM (one WG per (dir, 16-batch tile)) -------
// Per step: async-stage this step's 64KB xg tile into LDS while 32 WMMAs run on hS.
__global__ void __launch_bounds__(512)
k_lstm(const float* __restrict__ xg_f, const float* __restrict__ xg_b,
       const bf16_t* __restrict__ whh_f, const bf16_t* __restrict__ whh_b,
       const float* __restrict__ h0, const float* __restrict__ c0, int layer,
       const int* __restrict__ rev, bf16_t* __restrict__ out) {
    __shared__ __align__(16) bf16_t hS[16 * nH];       // 8 KB: h for this batch tile
    __shared__ __align__(16) float  xgS[16 * nG];      // 64 KB: current step xg tile
    __shared__ int stS[16];                            // per-row time index
    const int dir = blockIdx.x, mt = blockIdx.y;       // dir 0/1, batch tile 0..3
    const int tid = threadIdx.x, lane = tid & 31, wave = tid >> 5;   // 16 waves
    const float*  xg  = dir ? xg_b  : xg_f;
    const bf16_t* whh = dir ? whh_b : whh_f;
    const size_t  dof = (size_t)(layer * 2 + dir) * nB * nH + (size_t)mt * 16 * nH;
    const float*  h0d = h0 + dof;
    const float*  c0d = c0 + dof;

    for (int i = tid; i < 16 * nH; i += 512) hS[i] = f2bf(h0d[i]);

    const int hcol  = wave * 16 + (lane & 15);         // this wave's h-column patch
    const int lrow0 = (lane >> 4) << 3;                // 0 or 8 (C-layout rows)
    v8f c;
#pragma unroll
    for (int r = 0; r < 8; ++r) c[r] = c0d[(size_t)(lrow0 + r) * nH + hcol];
    __syncthreads();

    for (int t = 0; t < nT; ++t) {
        // ---- issue phase: wave w DMA-copies xg row for local batch row w ----
        {
            const int lb = wave;
            const int b  = mt * 16 + lb;
            const int st = dir ? rev[b * nT + t] : t;
            if (lane == 0) stS[lb] = st;
            const float* src = xg + ((size_t)b * nT + st) * nG;
#pragma unroll
            for (int it = 0; it < 8; ++it) {
                const int off = it * 128 + lane * 4;   // 4 floats / lane / issue
#if HAVE_ASYNC_LDS
                __builtin_amdgcn_global_load_async_to_lds_b128(
                    (as1_v4i)(src + off), (as3_v4i)(&xgS[lb * nG + off]), 0, 0);
#else
                *reinterpret_cast<v4f*>(&xgS[lb * nG + off]) =
                    *reinterpret_cast<const v4f*>(src + off);
#endif
            }
        }
        // ---- GEMM phase: gates += h_{t-1} * W_hh^T (overlaps with async DMA) ----
        v8f a0 = {}, a1 = {}, a2 = {}, a3 = {};
#pragma unroll
        for (int kk = 0; kk < nH; kk += 32) {
            v16bf a  = load_a_rm(hS, nH, kk, lane);
            v16bf b0 = load_b_rm(whh, nH, 0 * nH + hcol, kk, lane);
            v16bf b1 = load_b_rm(whh, nH, 1 * nH + hcol, kk, lane);
            v16bf b2 = load_b_rm(whh, nH, 2 * nH + hcol, kk, lane);
            v16bf b3 = load_b_rm(whh, nH, 3 * nH + hcol, kk, lane);
            a0 = wmma_bf16(a, b0, a0);
            a1 = wmma_bf16(a, b1, a1);
            a2 = wmma_bf16(a, b2, a2);
            a3 = wmma_bf16(a, b3, a3);
        }
        wait_asynccnt0();                              // this wave's DMA done
        __syncthreads();                               // hS reads done, xgS complete
        // ---- elementwise gate/cell update ----
#pragma unroll
        for (int r = 0; r < 8; ++r) {
            const int lb = lrow0 + r;
            const int b  = mt * 16 + lb;
            const float* xr = &xgS[lb * nG];
            float gi = xr[0 * nH + hcol] + a0[r];
            float gf = xr[1 * nH + hcol] + a1[r];
            float gg = xr[2 * nH + hcol] + a2[r];
            float go = xr[3 * nH + hcol] + a3[r];
            float cv = sigm(gf) * c[r] + sigm(gi) * tanhf(gg);
            float hv = sigm(go) * tanhf(cv);
            c[r] = cv;
            const int st = stS[lb];
            out[((size_t)b * nT + st) * nH2 + dir * nH + hcol] = f2bf(hv);
            hS[lb * nH + hcol] = f2bf(hv);
        }
        __syncthreads();                               // hS/xgS safe to reuse
    }
}

// ---------------- emission projection: em[M][48] = feats[M][512] * Wout[48][512]^T + b
__global__ void k_em(const bf16_t* __restrict__ feats, const bf16_t* __restrict__ W,
                     const float* __restrict__ b_out, const int* __restrict__ lengths,
                     float* __restrict__ em) {
    __shared__ bf16_t sA[16 * nH2];                    // 16 KB
    const int tid = threadIdx.x, lane = tid & 31, wave = tid >> 5;   // 3 waves
    const int mbase = blockIdx.x * 16;
    for (int i = tid; i < 16 * nH2; i += 96)
        sA[i] = feats[(size_t)(mbase + i / nH2) * nH2 + (i % nH2)];
    __syncthreads();
    const int n = wave * 16 + (lane & 15);
    v8f acc = {};
#pragma unroll
    for (int kk = 0; kk < nH2; kk += 32) {
        v16bf a = load_a_rm(sA, nH2, kk, lane);
        v16bf b = load_b_rm(W, nH2, n, kk, lane);
        acc = wmma_bf16(a, b, acc);
    }
    const float bv = (n < nLBL * nNT) ? b_out[n] : 0.f;
#pragma unroll
    for (int r = 0; r < 8; ++r) {
        int row = mbase + ((lane >> 4) << 3) + r;
        int b = row / nT, t = row % nT;
        bool valid = t < lengths[b];
        float v = (n < nLBL * nNT) ? ((valid ? acc[r] : 0.f) + bv) : 0.f;
        em[(size_t)row * nEMP + n] = v;
    }
}

// ---------------- CRF log-likelihood: one wave per batch element ----------------
__global__ void k_crf(const float* __restrict__ em, const int* __restrict__ y,
                      const float* __restrict__ stp, const float* __restrict__ enp,
                      const float* __restrict__ trp, float* __restrict__ partial) {
    __shared__ float tr[nNT * nNT], st[nNT], en[nNT];
    const int b = blockIdx.x, lane = threadIdx.x;
    for (int i = lane; i < nNT * nNT; i += 32) tr[i] = trp[i];
    if (lane < nNT) { st[lane] = stp[lane]; en[lane] = enp[lane]; }
    __syncthreads();
    const int j  = lane;
    const int jj = (j < nNT) ? j : 0;
    float alpha = (j < nNT) ? st[j] + emv(em, b, 0, j) : -1e30f;
    for (int s = 1; s < nS; ++s) {
        if (y[b * nS + s] == -1) continue;             // uniform over the wave
        float vals[nNT], mx = -1e30f;
#pragma unroll
        for (int i = 0; i < nNT; ++i) {
            float v = __shfl(alpha, i) + tr[i * nNT + jj];
            vals[i] = v; mx = fmaxf(mx, v);
        }
        float sum = 0.f;
#pragma unroll
        for (int i = 0; i < nNT; ++i) sum += __expf(vals[i] - mx);
        float nxt = mx + __logf(sum) + ((j < nNT) ? emv(em, b, s, j) : 0.f);
        if (j < nNT) alpha = nxt;
    }
    float term = (j < nNT) ? alpha + en[j] : -1e30f;
    float tv[nNT], mx = -1e30f;
#pragma unroll
    for (int i = 0; i < nNT; ++i) { tv[i] = __shfl(term, i); mx = fmaxf(mx, tv[i]); }
    float sum = 0.f;
#pragma unroll
    for (int i = 0; i < nNT; ++i) sum += __expf(tv[i] - mx);
    float logZ = mx + __logf(sum);
    if (lane == 0) {
        int y0 = y[b * nS];
        int tc = (y0 != -1) ? y0 : 0;
        float num = st[tc] + emv(em, b, 0, tc);
        int lastTc = tc, prev = tc;
        for (int s = 1; s < nS; ++s) {
            int ys = y[b * nS + s];
            int tcs = (ys != -1) ? ys : 0;
            if (ys != -1) { num += tr[prev * nNT + tcs] + emv(em, b, s, tcs); lastTc = tcs; }
            prev = tcs;
        }
        num += en[lastTc];
        partial[b] = num - logZ;
    }
}

__global__ void k_finalize(const float* __restrict__ partial, float* __restrict__ out) {
    if (threadIdx.x == 0) {
        float s = 0.f;
        for (int i = 0; i < nB; ++i) s += partial[i];
        out[0] = -s / (float)nB;
    }
}

// ---------------- Viterbi: one wave per batch element ----------------
__global__ void k_viterbi(const float* __restrict__ em, const int* __restrict__ y,
                          const float* __restrict__ stp, const float* __restrict__ enp,
                          const float* __restrict__ trp, unsigned char* __restrict__ hist,
                          float* __restrict__ dec) {
    __shared__ float tr[nNT * nNT], st[nNT], en[nNT];
    const int b = blockIdx.x, lane = threadIdx.x;
    for (int i = lane; i < nNT * nNT; i += 32) tr[i] = trp[i];
    if (lane < nNT) { st[lane] = stp[lane]; en[lane] = enp[lane]; }
    __syncthreads();
    const int j  = lane;
    const int jj = (j < nNT) ? j : 0;
    float score = (j < nNT) ? st[j] + emv(em, b, 0, j) : -1e30f;
    for (int s = 1; s < nS; ++s) {
        bool m = (y[b * nS + s] != -1);
        float best = -1e30f; int bi = 0;
#pragma unroll
        for (int i = 0; i < nNT; ++i) {
            float v = __shfl(score, i) + tr[i * nNT + jj];
            if (v > best) { best = v; bi = i; }
        }
        if (j < 16) hist[((size_t)b * nS + s) * 16 + j] = (unsigned char)bi;
        if (m && j < nNT) score = best + emv(em, b, s, j);
    }
    float term = (j < nNT) ? score + en[j] : -1e30f;
    float bv = -1e30f; int bj = 0;
#pragma unroll
    for (int i = 0; i < nNT; ++i) {
        float v = __shfl(term, i);
        if (v > bv) { bv = v; bj = i; }
    }
    if (lane == 0) {
        int carry = bj;
        dec[b * nS + (nS - 1)] = (float)carry;
        for (int s = nS - 1; s >= 1; --s) {
            int prev = hist[((size_t)b * nS + s) * 16 + carry];
            if (y[b * nS + s] != -1) carry = prev;
            dec[b * nS + (s - 1)] = (float)carry;
        }
    }
}

// ---------------- host launcher ----------------
extern "C" void kernel_launch(void* const* d_in, const int* in_sizes, int n_in,
                              void* d_out, int out_size, void* d_ws, size_t ws_size,
                              hipStream_t stream) {
    const float* x        = (const float*)d_in[0];
    const int*   y_true   = (const int*)  d_in[1];
    const int*   lengths  = (const int*)  d_in[2];
    const float* h0       = (const float*)d_in[3];
    const float* c0       = (const float*)d_in[4];
    const float* w_ih_l0  = (const float*)d_in[5];
    const float* w_hh_l0  = (const float*)d_in[6];
    const float* b_l0     = (const float*)d_in[7];
    const float* w_ih_l0r = (const float*)d_in[8];
    const float* w_hh_l0r = (const float*)d_in[9];
    const float* b_l0r    = (const float*)d_in[10];
    const float* w_ih_l1  = (const float*)d_in[11];
    const float* w_hh_l1  = (const float*)d_in[12];
    const float* b_l1     = (const float*)d_in[13];
    const float* w_ih_l1r = (const float*)d_in[14];
    const float* w_hh_l1r = (const float*)d_in[15];
    const float* b_l1r    = (const float*)d_in[16];
    const float* w_out    = (const float*)d_in[17];
    const float* b_out    = (const float*)d_in[18];
    const float* st       = (const float*)d_in[19];
    const float* en       = (const float*)d_in[20];
    const float* tr       = (const float*)d_in[21];

    char* cur = (char*)d_ws;
    auto alloc = [&](size_t bytes) -> char* {
        char* p = cur;
        cur += (bytes + 255) & ~(size_t)255;
        return p;
    };
    bf16_t* xt    = (bf16_t*)alloc((size_t)nB * nT * nF * 2);
    bf16_t* wih0f = (bf16_t*)alloc((size_t)nG * nF  * 2);
    bf16_t* wih0b = (bf16_t*)alloc((size_t)nG * nF  * 2);
    bf16_t* whh0f = (bf16_t*)alloc((size_t)nG * nH  * 2);
    bf16_t* whh0b = (bf16_t*)alloc((size_t)nG * nH  * 2);
    bf16_t* wih1f = (bf16_t*)alloc((size_t)nG * nH2 * 2);
    bf16_t* wih1b = (bf16_t*)alloc((size_t)nG * nH2 * 2);
    bf16_t* whh1f = (bf16_t*)alloc((size_t)nG * nH  * 2);
    bf16_t* whh1b = (bf16_t*)alloc((size_t)nG * nH  * 2);
    bf16_t* woutp = (bf16_t*)alloc((size_t)nEMP * nH2 * 2);
    float*  xg_f  = (float*)alloc((size_t)nB * nT * nG * 4);
    float*  xg_b  = (float*)alloc((size_t)nB * nT * nG * 4);
    bf16_t* l0    = (bf16_t*)alloc((size_t)nB * nT * nH2 * 2);
    bf16_t* l1    = (bf16_t*)alloc((size_t)nB * nT * nH2 * 2);
    float*  em    = (float*)alloc((size_t)nB * nT * nEMP * 4);
    int*    rev   = (int*)alloc((size_t)nB * nT * 4);
    unsigned char* hist = (unsigned char*)alloc((size_t)nB * nS * 16);
    float*  partial = (float*)alloc((size_t)nB * 4);

    const int M = nB * nT;                     // 65536 rows

    // prep
    k_transpose_x<<<(nB * nT * nF + 255) / 256, 256, 0, stream>>>(x, xt);
    k_cvt<<<(nG * nF  + 255) / 256, 256, 0, stream>>>(w_ih_l0,  wih0f, nG * nF);
    k_cvt<<<(nG * nF  + 255) / 256, 256, 0, stream>>>(w_ih_l0r, wih0b, nG * nF);
    k_cvt<<<(nG * nH  + 255) / 256, 256, 0, stream>>>(w_hh_l0,  whh0f, nG * nH);
    k_cvt<<<(nG * nH  + 255) / 256, 256, 0, stream>>>(w_hh_l0r, whh0b, nG * nH);
    k_cvt<<<(nG * nH2 + 255) / 256, 256, 0, stream>>>(w_ih_l1,  wih1f, nG * nH2);
    k_cvt<<<(nG * nH2 + 255) / 256, 256, 0, stream>>>(w_ih_l1r, wih1b, nG * nH2);
    k_cvt<<<(nG * nH  + 255) / 256, 256, 0, stream>>>(w_hh_l1,  whh1f, nG * nH);
    k_cvt<<<(nG * nH  + 255) / 256, 256, 0, stream>>>(w_hh_l1r, whh1b, nG * nH);
    k_cvt_pad_wout<<<(nEMP * nH2 + 255) / 256, 256, 0, stream>>>(w_out, woutp);
    k_rev<<<(nB * nT + 255) / 256, 256, 0, stream>>>(lengths, rev);

    // layer 0: input projection + recurrence
    k_xg_gemm<<<dim3(M / 16, 8), 256, 16 * nF * 2, stream>>>(xt, wih0f, b_l0,  xg_f, nF);
    k_xg_gemm<<<dim3(M / 16, 8), 256, 16 * nF * 2, stream>>>(xt, wih0b, b_l0r, xg_b, nF);
    k_lstm<<<dim3(2, 4), 512, 0, stream>>>(xg_f, xg_b, whh0f, whh0b, h0, c0, 0, rev, l0);

    // layer 1 (xg buffers reused)
    k_xg_gemm<<<dim3(M / 16, 8), 256, 16 * nH2 * 2, stream>>>(l0, wih1f, b_l1,  xg_f, nH2);
    k_xg_gemm<<<dim3(M / 16, 8), 256, 16 * nH2 * 2, stream>>>(l0, wih1b, b_l1r, xg_b, nH2);
    k_lstm<<<dim3(2, 4), 512, 0, stream>>>(xg_f, xg_b, whh1f, whh1b, h0, c0, 1, rev, l1);

    // emissions + CRF
    k_em<<<M / 16, 96, 0, stream>>>(l1, woutp, b_out, lengths, em);
    k_crf<<<nB, 32, 0, stream>>>(em, y_true, st, en, tr, partial);
    k_finalize<<<1, 32, 0, stream>>>(partial, (float*)d_out);
    k_viterbi<<<nB, 32, 0, stream>>>(em, y_true, st, en, tr, hist, (float*)d_out + 1);
}
